// SpatioTemporalLayer_55035710931607
// MI455X (gfx1250) — matrix operand order repaired and checked by reading, not compiled, not run on hardware
//
#include <hip/hip_runtime.h>

typedef __attribute__((ext_vector_type(16))) __bf16 v16bf;
typedef __attribute__((ext_vector_type(8)))  __bf16 v8bf;
typedef __attribute__((ext_vector_type(8)))  float  v8f;

#define N_   8
#define C_   256
#define T_   120
#define V_   25
#define W_   5
#define PAD_ 2
#define H_   8
#define CO_  256
#define HD_  32
#define L_   125
#define LP_  128
#define B_   (N_ * T_)        // 960 windows
#define CO3_ (3 * CO_)        // 768
#define P_   (N_ * T_ * V_)   // 24000 spatial positions
#define EPSF 1e-5f

// ---------- helpers ----------

__device__ __forceinline__ unsigned short f2bf_u(float f) {
  unsigned u = __builtin_bit_cast(unsigned, f);
  unsigned r = u + 0x7FFFu + ((u >> 16) & 1u);      // round-to-nearest-even
  return (unsigned short)(r >> 16);
}

__device__ __forceinline__ __bf16 f2bf(float f) {
  return __builtin_bit_cast(__bf16, f2bf_u(f));
}

// A fragment: 16x32 (MxK) bf16 from row-major, ld in elements (ld*2 and base 16B-aligned).
// lanes 0-15: row m=lane, K 0..15 ; lanes 16-31: row m=lane-16, K 16..31
__device__ __forceinline__ v16bf load_a(const __bf16* p, int ld) {
  int lane = threadIdx.x & 31;
  int m  = lane & 15;
  int kb = (lane >> 4) << 4;          // 0 or 16
  const __bf16* r = p + m * ld + kb;
  v8bf lo = *(const v8bf*)r;          // 16B
  v8bf hi = *(const v8bf*)(r + 8);    // 16B
  return __builtin_shufflevector(lo, hi, 0,1,2,3,4,5,6,7,8,9,10,11,12,13,14,15);
}

// B fragment where stored matrix is row-major N x K (B = stored^T); contiguous in K.
// lanes 0-15: col n=lane, K 0..15 ; lanes 16-31: col n=lane-16, K 16..31
__device__ __forceinline__ v16bf load_bt(const __bf16* p, int ld) {
  int lane = threadIdx.x & 31;
  int n  = lane & 15;
  int kb = (lane >> 4) << 4;          // 0 or 16
  const __bf16* r = p + n * ld + kb;
  v8bf lo = *(const v8bf*)r;
  v8bf hi = *(const v8bf*)(r + 8);
  return __builtin_shufflevector(lo, hi, 0,1,2,3,4,5,6,7,8,9,10,11,12,13,14,15);
}

// C/D 16x16 f32: lanes 0-15 -> (M=r, N=lane); lanes 16-31 -> (M=8+r, N=lane-16)
__device__ __forceinline__ void store_c(float* p, int ld, v8f c) {
  int lane = threadIdx.x & 31;
  int n  = lane & 15;
  int mb = (lane >> 4) << 3;
#pragma unroll
  for (int r = 0; r < 8; ++r) p[(mb + r) * ld + n] = c[r];
}

#define WMMA_BF16(a, b, c) \
  __builtin_amdgcn_wmma_f32_16x16x32_bf16(false, (a), false, (b), (short)0, (c), false, false)

// Fill a 256x64 B panel into LDS in WMMA *fragment-major* layout:
// fragment f = kk*4+j (kk=0..7 k-step, j=0..3 col tile); per fragment 32 lanes x 16 bf16
// contiguous, so the GEMM loop loads one aligned 32B chunk per lane (2x ds_load_b128).
// transposed=false: B[k][col] = src[k*src_ld + col0+col]
// transposed=true : B[k][col] = src[(col0+col)*src_ld + k]   (k-contiguous reads)
template <bool TRANSPOSED>
__device__ __forceinline__ void fill_panel(__bf16* Bs, const __bf16* src, int src_ld, int col0) {
  for (int g = threadIdx.x; g < 1024; g += 256) {   // 32 fragments x 32 lanes
    int f    = g >> 5;
    int lane = g & 31;
    int kk = f >> 2, j = f & 3;
    int kbase = kk * 32 + ((lane >> 4) << 4);
    int col   = col0 + j * 16 + (lane & 15);
    v16bf tmp;
#pragma unroll
    for (int i = 0; i < 16; ++i) {
      int k = kbase + i;
      tmp[i] = TRANSPOSED ? src[(size_t)col * src_ld + k]
                          : src[(size_t)k * src_ld + col];
    }
    *(v16bf*)(Bs + (size_t)g * 16) = tmp;
  }
}

// ---------- kernels ----------

__global__ void k_f2bf(const float* __restrict__ src, __bf16* __restrict__ dst, int n) {
  int i = blockIdx.x * 256 + threadIdx.x;
  if (i < n) dst[i] = f2bf(src[i]);
}

// Build xg[b][l][c] (B_,128,256) bf16; l = w*25+v, rows 125..127 zero, OOB time zero.
__global__ void k_gather(const float* __restrict__ x, __bf16* __restrict__ xg) {
  int idx = blockIdx.x * 256 + threadIdx.x;
  int c = idx & (C_ - 1);
  int l = (idx >> 8) & (LP_ - 1);
  int b = idx >> 15;
  float val = 0.f;
  if (l < L_) {
    int w  = l / V_, vv = l % V_;
    int n  = b / T_, t  = b % T_;
    int tt = t + w - PAD_;
    if (tt >= 0 && tt < T_)
      val = x[(((size_t)n * C_ + c) * T_ + tt) * V_ + vv];
  }
  xg[idx] = f2bf(val);
}

// qkv = xg(128x256) @ w_qkv(256x768); block = (window, 64-col group), swizzled B panel in LDS.
// V is written transposed: Vt (B,H,HD,LP) so the attn AV GEMM gets contiguous B fragments.
__global__ __launch_bounds__(256) void k_qkv(const __bf16* __restrict__ xg,
                                             const __bf16* __restrict__ wq,
                                             __bf16* __restrict__ Qb,
                                             __bf16* __restrict__ Kb,
                                             __bf16* __restrict__ Vt) {
  __shared__ __bf16 Bs[C_ * 64];                  // 32 KB, fragment-major
  int b  = blockIdx.x / 12;
  int cg = blockIdx.x % 12;
  fill_panel<false>(Bs, wq, CO3_, cg * 64);
  __syncthreads();
  int rt   = threadIdx.x >> 5;                    // wave -> row tile 0..7
  int lane = threadIdx.x & 31;
  const __bf16* A = xg + ((size_t)b * LP_ + rt * 16) * C_;
  const v16bf* Bf = (const v16bf*)Bs;
  v8f acc[4] = {};
#pragma unroll
  for (int kk = 0; kk < C_ / 32; ++kk) {
    v16bf a = load_a(A + kk * 32, C_);
#pragma unroll
    for (int j = 0; j < 4; ++j)
      acc[j] = WMMA_BF16(a, Bf[(kk * 4 + j) * 32 + lane], acc[j]);
  }
  int n  = lane & 15;
  int mb = (lane >> 4) << 3;
#pragma unroll
  for (int j = 0; j < 4; ++j) {
    int o = cg * 64 + j * 16 + n;                 // 0..767
    int h = o / 96;
    int s = (o % 96) / 32;                        // 0=q 1=k 2=v
    int d = o % 32;
    if (s == 2) {                                 // V transposed: contiguous in l
      __bf16* dv = Vt + (((size_t)b * H_ + h) * HD_ + d) * LP_ + rt * 16 + mb;
#pragma unroll
      for (int r = 0; r < 8; ++r) dv[r] = f2bf(acc[j][r]);
    } else {
      __bf16* dst = (s == 0) ? Qb : Kb;
      float sc = (s == 0) ? 0.0625f : 1.0f;       // fold 1/sqrt(CO) into Q
#pragma unroll
      for (int r = 0; r < 8; ++r) {
        int l = rt * 16 + mb + r;
        dst[(((size_t)b * H_ + h) * LP_ + l) * HD_ + d] = f2bf(acc[j][r] * sc);
      }
    }
  }
}

// Fused attention per (window, head). Scores in 64KB LDS f32, masked softmax (float4),
// bf16 overlay in-place, attn @ V with transposed-V contiguous fragments.
__global__ __launch_bounds__(256) void k_attn(const __bf16* __restrict__ Qb,
                                              const __bf16* __restrict__ Kb,
                                              const __bf16* __restrict__ Vt,
                                              __bf16* __restrict__ AO) {
  __shared__ float sc[LP_ * LP_];                 // 64 KB
  __bf16* sa = (__bf16*)sc;                       // bf16 overlay, row stride 256 (same 512B rows)
  int b  = blockIdx.x >> 3;
  int h  = blockIdx.x & 7;
  int wv = threadIdx.x >> 5;
  const __bf16* Qh  = Qb + ((size_t)b * H_ + h) * LP_ * HD_;
  const __bf16* Kh  = Kb + ((size_t)b * H_ + h) * LP_ * HD_;
  const __bf16* Vth = Vt + ((size_t)b * H_ + h) * HD_ * LP_;   // (HD, LP)

  v16bf aq = load_a(Qh + wv * 16 * HD_, HD_);     // K = HD = 32: one WMMA per score tile
#pragma unroll
  for (int ct = 0; ct < 8; ++ct) {
    v16bf bk = load_bt(Kh + ct * 16 * HD_, HD_);  // B = K^T, contiguous
    v8f s0 = {};
    s0 = WMMA_BF16(aq, bk, s0);
    store_c(sc + wv * 16 * LP_ + ct * 16, LP_, s0);
  }
  __syncthreads();
  if (threadIdx.x < LP_) {                        // masked softmax, row per thread
    float* row = sc + threadIdx.x * LP_;
    row[125] = -1e30f; row[126] = -1e30f; row[127] = -1e30f;   // mask pad columns
    const float4* row4 = (const float4*)row;
    float mx = -1e30f;
#pragma unroll 4
    for (int c = 0; c < 32; ++c) {
      float4 q = row4[c];
      mx = fmaxf(mx, fmaxf(fmaxf(q.x, q.y), fmaxf(q.z, q.w)));
    }
    float sum = 0.f;
#pragma unroll 4
    for (int c = 0; c < 32; ++c) {
      float4 q = row4[c];
      sum += __expf(q.x - mx) + __expf(q.y - mx) + __expf(q.z - mx) + __expf(q.w - mx);
    }
    float inv = 1.f / sum;
    unsigned long long* arow = (unsigned long long*)(sa + threadIdx.x * 256);
#pragma unroll 4
    for (int c = 0; c < 32; ++c) {                // write 4 bf16 as one 8B store
      float4 q = row4[c];                         // read 16B at 16c, write 8B at 8c: safe
      unsigned long long pk =
          (unsigned long long)f2bf_u(__expf(q.x - mx) * inv)
        | ((unsigned long long)f2bf_u(__expf(q.y - mx) * inv) << 16)
        | ((unsigned long long)f2bf_u(__expf(q.z - mx) * inv) << 32)
        | ((unsigned long long)f2bf_u(__expf(q.w - mx) * inv) << 48);
      arow[c] = pk;
    }
  }
  __syncthreads();
  v8f o0 = {}, o1 = {};                           // wave: row tile wv, cols 0..31
#pragma unroll
  for (int kk = 0; kk < 4; ++kk) {
    v16bf a  = load_a(sa + wv * 16 * 256 + kk * 32, 256);
    v16bf b0 = load_bt(Vth + (size_t) 0 * LP_ + kk * 32, LP_);
    v16bf b1 = load_bt(Vth + (size_t)16 * LP_ + kk * 32, LP_);
    o0 = WMMA_BF16(a, b0, o0);
    o1 = WMMA_BF16(a, b1, o1);
  }
  int lane = threadIdx.x & 31;
  int n  = lane & 15;
  int mb = (lane >> 4) << 3;
#pragma unroll
  for (int r = 0; r < 8; ++r) {
    int l = wv * 16 + mb + r;
    size_t base = ((size_t)b * LP_ + l) * CO_ + h * HD_;
    AO[base + n]      = f2bf(o0[r]);
    AO[base + 16 + n] = f2bf(o1[r]);
  }
}

// out2 = AO(122880x256) @ w_out + b_out, f32 output.
__global__ __launch_bounds__(256) void k_outproj(const __bf16* __restrict__ AO,
                                                 const __bf16* __restrict__ wo,
                                                 const float* __restrict__ bo,
                                                 float* __restrict__ out2) {
  __shared__ __bf16 Bs[CO_ * 64];
  int rtg = blockIdx.x >> 2;
  int cg  = blockIdx.x & 3;
  fill_panel<false>(Bs, wo, CO_, cg * 64);
  __syncthreads();
  int rt   = rtg * 8 + (threadIdx.x >> 5);        // < 7680
  int lane = threadIdx.x & 31;
  const __bf16* A = AO + (size_t)rt * 16 * CO_;
  const v16bf* Bf = (const v16bf*)Bs;
  v8f acc[4] = {};
#pragma unroll
  for (int kk = 0; kk < CO_ / 32; ++kk) {
    v16bf a = load_a(A + kk * 32, CO_);
#pragma unroll
    for (int j = 0; j < 4; ++j)
      acc[j] = WMMA_BF16(a, Bf[(kk * 4 + j) * 32 + lane], acc[j]);
  }
  int n  = lane & 15;
  int mb = (lane >> 4) << 3;
#pragma unroll
  for (int j = 0; j < 4; ++j) {
    int o = cg * 64 + j * 16 + n;
    float bv = bo[o];
#pragma unroll
    for (int r = 0; r < 8; ++r)
      out2[((size_t)rt * 16 + mb + r) * CO_ + o] = acc[j][r] + bv;
  }
}

// mean over W, residual, BN1; write f32 (N,CO,T,V) and bf16 (P,256) row-major for FFN.
__global__ void k_mean_bn1(const float* __restrict__ out2, const float* __restrict__ x,
                           const float* __restrict__ g, const float* __restrict__ bb,
                           const float* __restrict__ mm, const float* __restrict__ vv,
                           float* __restrict__ inter, __bf16* __restrict__ XT) {
  int idx = blockIdx.x * 256 + threadIdx.x;       // < N*CO*T*V
  int v  = idx % V_;
  int t  = (idx / V_) % T_;
  int co = (idx / (V_ * T_)) % CO_;
  int n  = idx / (V_ * T_ * CO_);
  size_t wbase = (size_t)(n * T_ + t) * LP_;
  float s = 0.f;
#pragma unroll
  for (int w = 0; w < W_; ++w)
    s += out2[(wbase + w * V_ + v) * CO_ + co];
  float y = s * 0.2f + x[idx];
  y = (y - mm[co]) * rsqrtf(vv[co] + EPSF) * g[co] + bb[co];
  inter[idx] = y;
  size_t p = (size_t)(n * T_ + t) * V_ + v;
  XT[p * CO_ + co] = f2bf(y);
}

// h1 = BN_ffn(mish(X @ W1^T)), bf16 output (P,256).
__global__ __launch_bounds__(256) void k_ffn1(const __bf16* __restrict__ X,
                                              const __bf16* __restrict__ w1,
                                              const float* __restrict__ g, const float* __restrict__ bb,
                                              const float* __restrict__ mm, const float* __restrict__ vv,
                                              __bf16* __restrict__ Y) {
  __shared__ __bf16 Bs[CO_ * 64];
  int rtg = blockIdx.x >> 2;
  int cg  = blockIdx.x & 3;
  fill_panel<true>(Bs, w1, CO_, cg * 64);         // B = W1^T, k-contiguous fill
  __syncthreads();
  int rt = rtg * 8 + (threadIdx.x >> 5);
  if (rt >= P_ / 16) return;                      // whole waves exit together
  int lane = threadIdx.x & 31;
  const __bf16* A = X + (size_t)rt * 16 * CO_;
  const v16bf* Bf = (const v16bf*)Bs;
  v8f acc[4] = {};
#pragma unroll
  for (int kk = 0; kk < CO_ / 32; ++kk) {
    v16bf a = load_a(A + kk * 32, CO_);
#pragma unroll
    for (int j = 0; j < 4; ++j)
      acc[j] = WMMA_BF16(a, Bf[(kk * 4 + j) * 32 + lane], acc[j]);
  }
  int n  = lane & 15;
  int mb = (lane >> 4) << 3;
#pragma unroll
  for (int j = 0; j < 4; ++j) {
    int o = cg * 64 + j * 16 + n;
    float inv = rsqrtf(vv[o] + EPSF) * g[o];
    float mo = mm[o], bo = bb[o];
#pragma unroll
    for (int r = 0; r < 8; ++r) {
      float val = acc[j][r];
      float sp  = (val > 20.f) ? val : log1pf(__expf(val));   // softplus
      float y   = val * tanhf(sp);                            // mish
      Y[((size_t)rt * 16 + mb + r) * CO_ + o] = f2bf((y - mo) * inv + bo);
    }
  }
}

// out = BN2(h1 @ W2^T + inter), scattered into (N,CO,T,V) f32.
__global__ __launch_bounds__(256) void k_ffn2(const __bf16* __restrict__ Hin,
                                              const __bf16* __restrict__ w2,
                                              const float* __restrict__ inter,
                                              const float* __restrict__ g, const float* __restrict__ bb,
                                              const float* __restrict__ mm, const float* __restrict__ vv,
                                              float* __restrict__ out) {
  __shared__ __bf16 Bs[CO_ * 64];
  int rtg = blockIdx.x >> 2;
  int cg  = blockIdx.x & 3;
  fill_panel<true>(Bs, w2, CO_, cg * 64);
  __syncthreads();
  int rt = rtg * 8 + (threadIdx.x >> 5);
  if (rt >= P_ / 16) return;
  int lane = threadIdx.x & 31;
  const __bf16* A = Hin + (size_t)rt * 16 * CO_;
  const v16bf* Bf = (const v16bf*)Bs;
  v8f acc[4] = {};
#pragma unroll
  for (int kk = 0; kk < CO_ / 32; ++kk) {
    v16bf a = load_a(A + kk * 32, CO_);
#pragma unroll
    for (int j = 0; j < 4; ++j)
      acc[j] = WMMA_BF16(a, Bf[(kk * 4 + j) * 32 + lane], acc[j]);
  }
  int n  = lane & 15;
  int mb = (lane >> 4) << 3;
#pragma unroll
  for (int j = 0; j < 4; ++j) {
    int o = cg * 64 + j * 16 + n;
    float inv = rsqrtf(vv[o] + EPSF) * g[o];
    float mo = mm[o], bo = bb[o];
#pragma unroll
    for (int r = 0; r < 8; ++r) {
      int p  = rt * 16 + mb + r;
      int nn = p / (T_ * V_);
      int rr = p % (T_ * V_);
      int t  = rr / V_;
      int v  = rr % V_;
      size_t idx = (((size_t)nn * CO_ + o) * T_ + t) * V_ + v;
      float val = acc[j][r] + inter[idx];
      out[idx] = (val - mo) * inv + bo;
    }
  }
}

// ---------- launch ----------

extern "C" void kernel_launch(void* const* d_in, const int* in_sizes, int n_in,
                              void* d_out, int out_size, void* d_ws, size_t ws_size,
                              hipStream_t stream) {
  const float* x      = (const float*)d_in[0];
  const float* w_qkv  = (const float*)d_in[1];
  const float* w_out  = (const float*)d_in[2];
  const float* b_out  = (const float*)d_in[3];
  const float* bn1_g  = (const float*)d_in[4];
  const float* bn1_b  = (const float*)d_in[5];
  const float* ffn_w1 = (const float*)d_in[6];
  const float* ffn_w2 = (const float*)d_in[7];
  const float* ffn_g  = (const float*)d_in[8];
  const float* ffn_b  = (const float*)d_in[9];
  const float* bn2_g  = (const float*)d_in[10];
  const float* bn2_b  = (const float*)d_in[11];
  const float* bn1_m  = (const float*)d_in[12];
  const float* bn1_v  = (const float*)d_in[13];
  const float* ffn_m  = (const float*)d_in[14];
  const float* ffn_v  = (const float*)d_in[15];
  const float* bn2_m  = (const float*)d_in[16];
  const float* bn2_v  = (const float*)d_in[17];
  float* out = (float*)d_out;

  char* base = (char*)d_ws;
  size_t off = 0;
  auto alloc = [&](size_t bytes) -> void* {
    void* p = base + off;
    off = (off + bytes + 255) & ~(size_t)255;
    return p;
  };
  __bf16* wq_bf = (__bf16*)alloc((size_t)C_ * CO3_ * 2);
  __bf16* wo_bf = (__bf16*)alloc((size_t)CO_ * CO_ * 2);
  __bf16* w1_bf = (__bf16*)alloc((size_t)CO_ * CO_ * 2);
  __bf16* w2_bf = (__bf16*)alloc((size_t)CO_ * CO_ * 2);
  __bf16* xg    = (__bf16*)alloc((size_t)B_ * LP_ * C_ * 2);
  __bf16* Qb    = (__bf16*)alloc((size_t)B_ * H_ * LP_ * HD_ * 2);
  __bf16* Kb    = (__bf16*)alloc((size_t)B_ * H_ * LP_ * HD_ * 2);
  __bf16* Vt    = (__bf16*)alloc((size_t)B_ * H_ * HD_ * LP_ * 2);
  __bf16* AO    = (__bf16*)alloc((size_t)B_ * LP_ * CO_ * 2);
  float*  out2  = (float*) alloc((size_t)B_ * LP_ * CO_ * 4);
  float*  inter = (float*) alloc((size_t)N_ * CO_ * T_ * V_ * 4);
  __bf16* XT    = (__bf16*)alloc((size_t)P_ * CO_ * 2);
  __bf16* H1    = (__bf16*)alloc((size_t)P_ * CO_ * 2);

  k_f2bf<<<(C_ * CO3_ + 255) / 256, 256, 0, stream>>>(w_qkv,  wq_bf, C_ * CO3_);
  k_f2bf<<<(CO_ * CO_ + 255) / 256, 256, 0, stream>>>(w_out,  wo_bf, CO_ * CO_);
  k_f2bf<<<(CO_ * CO_ + 255) / 256, 256, 0, stream>>>(ffn_w1, w1_bf, CO_ * CO_);
  k_f2bf<<<(CO_ * CO_ + 255) / 256, 256, 0, stream>>>(ffn_w2, w2_bf, CO_ * CO_);

  k_gather<<<(B_ * LP_ * C_) / 256, 256, 0, stream>>>(x, xg);
  k_qkv<<<B_ * 12, 256, 0, stream>>>(xg, wq_bf, Qb, Kb, Vt);
  k_attn<<<B_ * H_, 256, 0, stream>>>(Qb, Kb, Vt, AO);
  k_outproj<<<(B_ * LP_ / 16 / 8) * 4, 256, 0, stream>>>(AO, wo_bf, b_out, out2);
  k_mean_bn1<<<(N_ * CO_ * T_ * V_) / 256, 256, 0, stream>>>(out2, x, bn1_g, bn1_b, bn1_m, bn1_v, inter, XT);
  k_ffn1<<<188 * 4, 256, 0, stream>>>(XT, w1_bf, ffn_g, ffn_b, ffn_m, ffn_v, H1);
  k_ffn2<<<188 * 4, 256, 0, stream>>>(H1, w2_bf, inter, bn2_g, bn2_b, bn2_m, bn2_v, out);
}
